// TGNPLMemory_33174327394705
// MI455X (gfx1250) — matrix operand order repaired and checked by reading, not compiled, not run on hardware
//
#include <hip/hip_runtime.h>
#include <hip/hip_bf16.h>

// ---------------- problem constants ----------------
#define NNODES   100000
#define E_EDGES  50000
#define STATE_D  128
#define MEM_D    384      // STATE_DIM + NUM_PRODS
#define RAW_D    64
#define TIME_D   64
#define MSG_D    1280     // 3*384 + 64 + 64
#define GATE3    384      // 3 * STATE_DIM
#define ROWS     48       // edge rows / nodes per block (3 row-tiles of 16)
#define RT_TILES 3        // ROWS / 16

typedef __attribute__((ext_vector_type(16))) __bf16 v16bf;
typedef __attribute__((ext_vector_type(8)))  __bf16 v8bf;
typedef __attribute__((ext_vector_type(4)))  __bf16 v4bf;
typedef __attribute__((ext_vector_type(8)))  float  v8f;

union BFrag { v16bf v; v8bf h[2]; };

__device__ __forceinline__ v8f wmma_bf16(v16bf a, v16bf b, v8f c) {
    return __builtin_amdgcn_wmma_f32_16x16x32_bf16(
        false, a, false, b, (short)0, c, false, false);
}

// A fragment: 16 contiguous bf16 along K at p (matches B fragment k order)
__device__ __forceinline__ v16bf load_afrag(const __bf16* p) {
    BFrag a;
    a.h[0] = *(const v8bf*)p;
    a.h[1] = *(const v8bf*)(p + 8);
    return a.v;
}
// B fragment: 16 contiguous bf16 along K for one output column
__device__ __forceinline__ v16bf load_bfrag(const __bf16* p) {
    BFrag b;
    b.h[0] = *(const v8bf*)p;
    b.h[1] = *(const v8bf*)(p + 8);
    return b.v;
}

// ---------------- kernel 0a: zero accumulators ----------------
__global__ __launch_bounds__(256) void k_init(float* gi_acc, float* cnt, int* lu) {
    size_t i = (size_t)blockIdx.x * blockDim.x + threadIdx.x;
    size_t stride = (size_t)gridDim.x * blockDim.x;
    const size_t n_gi = (size_t)NNODES * GATE3;
    for (size_t j = i; j < n_gi; j += stride) gi_acc[j] = 0.0f;
    for (size_t j = i; j < NNODES; j += stride) { cnt[j] = 0.0f; lu[j] = 0; }
}

// ---------------- kernel 0b: f32 -> bf16 weight convert ----------------
__global__ __launch_bounds__(256) void k_convert(const float* __restrict__ in,
                                                 __bf16* __restrict__ out, int n) {
    int i = blockIdx.x * blockDim.x + threadIdx.x;
    int stride = gridDim.x * blockDim.x;
    for (int j = i; j < n; j += stride) out[j] = (__bf16)in[j];
}

// ---------------- kernel 1: per-edge message GEMM + scatter ----------------
// grid = ceil(E/48), 256 threads (8 waves). Each block: 48 edges (3 row-tiles).
__global__ __launch_bounds__(256) void k_edge_gemm(
    const int* __restrict__ src, const int* __restrict__ dst,
    const int* __restrict__ prod, const int* __restrict__ t,
    const float* __restrict__ raw_msg, const float* __restrict__ memory,
    const int* __restrict__ last_update,
    const float* __restrict__ w_time, const float* __restrict__ b_time,
    const __bf16* __restrict__ w_ih_bf,
    float* __restrict__ gi_acc, float* __restrict__ cnt, int* __restrict__ lu_max)
{
    __shared__ __align__(16) __bf16 lds_msg[ROWS][MSG_D];   // 120 KB
    __shared__ int   s_src[ROWS], s_dst[ROWS], s_prod[ROWS], s_t[ROWS];
    __shared__ float s_trel[ROWS];
    __shared__ float s_wt[TIME_D], s_bt[TIME_D];

    const int tid  = threadIdx.x;
    const int base = blockIdx.x * ROWS;

    if (tid < TIME_D) { s_wt[tid] = w_time[tid]; s_bt[tid] = b_time[tid]; }
    if (tid < ROWS) {
        int e = base + tid;
        int sv = 0, dv = 0, pv = 0, tv = 0; float tr = 0.0f;
        if (e < E_EDGES) {
            sv = src[e]; dv = dst[e]; pv = prod[e]; tv = t[e];
            tr = (float)(tv - last_update[sv]);
        }
        s_src[tid] = sv; s_dst[tid] = dv; s_prod[tid] = pv;
        s_t[tid] = tv; s_trel[tid] = tr;
    }
    __syncthreads();

    // vectorized gather + time encoding -> bf16 message tile in LDS
    // float4 loads (all segment boundaries are multiples of 4 cols)
    for (int i = tid; i < ROWS * (MSG_D / 4); i += 256) {
        int row = i / (MSG_D / 4);
        int col = (i - row * (MSG_D / 4)) * 4;
        int e = base + row;
        float4 v = make_float4(0.f, 0.f, 0.f, 0.f);
        if (e < E_EDGES) {
            if (col < 384)
                v = *(const float4*)&memory[(size_t)s_src[row]  * MEM_D + col];
            else if (col < 768)
                v = *(const float4*)&memory[(size_t)s_dst[row]  * MEM_D + (col - 384)];
            else if (col < 1152)
                v = *(const float4*)&memory[(size_t)s_prod[row] * MEM_D + (col - 768)];
            else if (col < 1216)
                v = *(const float4*)&raw_msg[(size_t)e * RAW_D + (col - 1152)];
            else {
                int j = col - 1216;
                float tr = s_trel[row];
                v.x = __cosf(tr * s_wt[j]     + s_bt[j]);
                v.y = __cosf(tr * s_wt[j + 1] + s_bt[j + 1]);
                v.z = __cosf(tr * s_wt[j + 2] + s_bt[j + 2]);
                v.w = __cosf(tr * s_wt[j + 3] + s_bt[j + 3]);
            }
        }
        v4bf o;
        o[0] = (__bf16)v.x; o[1] = (__bf16)v.y; o[2] = (__bf16)v.z; o[3] = (__bf16)v.w;
        *(v4bf*)&lds_msg[row][col] = o;
    }

    // per-edge count / timestamp scatter
    if (tid < ROWS) {
        int e = base + tid;
        if (e < E_EDGES) {
            atomicAdd(&cnt[s_src[tid]],  1.0f);
            atomicAdd(&cnt[s_dst[tid]],  1.0f);
            atomicAdd(&cnt[s_prod[tid]], 1.0f);
            atomicMax(&lu_max[s_src[tid]],  s_t[tid]);
            atomicMax(&lu_max[s_dst[tid]],  s_t[tid]);
            atomicMax(&lu_max[s_prod[tid]], s_t[tid]);
        }
    }
    __syncthreads();

    // WMMA: 48x384 output per block; wave w does columns [48w, 48w+48)
    const int lane = tid & 31;
    const int wave = tid >> 5;
    const int m  = lane & 15;
    const int hi = lane >> 4;
    const int nb0 = wave * 48;

    v8f acc[RT_TILES][3];
    #pragma unroll
    for (int rt = 0; rt < RT_TILES; ++rt)
        #pragma unroll
        for (int c = 0; c < 3; ++c) acc[rt][c] = (v8f){};

    const __bf16* b0p = w_ih_bf + (size_t)(nb0 +  0 + m) * MSG_D + (hi << 4);
    const __bf16* b1p = w_ih_bf + (size_t)(nb0 + 16 + m) * MSG_D + (hi << 4);
    const __bf16* b2p = w_ih_bf + (size_t)(nb0 + 32 + m) * MSG_D + (hi << 4);

    for (int k0 = 0; k0 < MSG_D; k0 += 32) {
        // 3 B fragments feed 9 WMMAs (3 row-tiles)
        v16bf b0 = load_bfrag(b0p + k0);
        v16bf b1 = load_bfrag(b1p + k0);
        v16bf b2 = load_bfrag(b2p + k0);
        const int kLo = k0 + (hi << 4);
        #pragma unroll
        for (int rt = 0; rt < RT_TILES; ++rt) {
            v16bf a = load_afrag(&lds_msg[rt * 16 + m][kLo]);
            acc[rt][0] = wmma_bf16(a, b0, acc[rt][0]);
            acc[rt][1] = wmma_bf16(a, b1, acc[rt][1]);
            acc[rt][2] = wmma_bf16(a, b2, acc[rt][2]);
        }
    }

    // scatter y_e into node accumulators (src, dst, prod each get a copy)
    #pragma unroll
    for (int rt = 0; rt < RT_TILES; ++rt) {
        #pragma unroll
        for (int r = 0; r < 8; ++r) {
            int row = rt * 16 + r + (hi << 3);
            int e = base + row;
            if (e < E_EDGES) {
                size_t bs = (size_t)s_src[row]  * GATE3;
                size_t bd = (size_t)s_dst[row]  * GATE3;
                size_t bp = (size_t)s_prod[row] * GATE3;
                int col0 = nb0 + m;
                float v0 = acc[rt][0][r], v1 = acc[rt][1][r], v2 = acc[rt][2][r];
                atomicAdd(&gi_acc[bs + col0],      v0);
                atomicAdd(&gi_acc[bd + col0],      v0);
                atomicAdd(&gi_acc[bp + col0],      v0);
                atomicAdd(&gi_acc[bs + col0 + 16], v1);
                atomicAdd(&gi_acc[bd + col0 + 16], v1);
                atomicAdd(&gi_acc[bp + col0 + 16], v1);
                atomicAdd(&gi_acc[bs + col0 + 32], v2);
                atomicAdd(&gi_acc[bd + col0 + 32], v2);
                atomicAdd(&gi_acc[bp + col0 + 32], v2);
            }
        }
    }
}

// ---------------- kernel 2: GRU over all nodes ----------------
// grid = ceil(N/48), 256 threads (8 waves). Each block: 48 nodes (3 row-tiles).
__global__ __launch_bounds__(256) void k_gru(
    const float* __restrict__ memory, const __bf16* __restrict__ w_hh_bf,
    const float* __restrict__ b_ih, const float* __restrict__ b_hh,
    const float* __restrict__ gi_acc, const float* __restrict__ cnt,
    const int* __restrict__ lu_max,
    float* __restrict__ out_mem, float* __restrict__ out_lu)
{
    __shared__ __align__(16) __bf16 lds_h[ROWS][STATE_D];  // 12 KB
    __shared__ float lds_gh[ROWS][GATE3];                  // 72 KB

    const int tid  = threadIdx.x;
    const int base = blockIdx.x * ROWS;

    for (int i = tid; i < ROWS * STATE_D; i += 256) {
        int row = i >> 7, c = i & 127;
        int node = base + row;
        float v = (node < NNODES) ? memory[(size_t)node * MEM_D + c] : 0.0f;
        lds_h[row][c] = (__bf16)v;
    }
    __syncthreads();

    const int lane = tid & 31;
    const int wave = tid >> 5;
    const int m  = lane & 15;
    const int hi = lane >> 4;
    const int nb0 = wave * 48;

    v8f acc[RT_TILES][3];
    #pragma unroll
    for (int rt = 0; rt < RT_TILES; ++rt)
        #pragma unroll
        for (int c = 0; c < 3; ++c) acc[rt][c] = (v8f){};

    const __bf16* b0p = w_hh_bf + (size_t)(nb0 +  0 + m) * STATE_D + (hi << 4);
    const __bf16* b1p = w_hh_bf + (size_t)(nb0 + 16 + m) * STATE_D + (hi << 4);
    const __bf16* b2p = w_hh_bf + (size_t)(nb0 + 32 + m) * STATE_D + (hi << 4);

    #pragma unroll
    for (int k0 = 0; k0 < STATE_D; k0 += 32) {
        v16bf b0 = load_bfrag(b0p + k0);
        v16bf b1 = load_bfrag(b1p + k0);
        v16bf b2 = load_bfrag(b2p + k0);
        const int kLo = k0 + (hi << 4);
        #pragma unroll
        for (int rt = 0; rt < RT_TILES; ++rt) {
            v16bf a = load_afrag(&lds_h[rt * 16 + m][kLo]);
            acc[rt][0] = wmma_bf16(a, b0, acc[rt][0]);
            acc[rt][1] = wmma_bf16(a, b1, acc[rt][1]);
            acc[rt][2] = wmma_bf16(a, b2, acc[rt][2]);
        }
    }

    #pragma unroll
    for (int rt = 0; rt < RT_TILES; ++rt) {
        #pragma unroll
        for (int r = 0; r < 8; ++r) {
            int row = rt * 16 + r + (hi << 3);
            lds_gh[row][nb0 + m]      = acc[rt][0][r];
            lds_gh[row][nb0 + 16 + m] = acc[rt][1][r];
            lds_gh[row][nb0 + 32 + m] = acc[rt][2][r];
        }
    }
    __syncthreads();

    // fused GRU gate math: gi = scatter_sum/cnt + b_ih ; gh = h@w_hh^T + b_hh
    for (int i = tid; i < ROWS * STATE_D; i += 256) {
        int row = i >> 7, o = i & 127;
        int node = base + row;
        if (node < NNODES) {
            float inv = 1.0f / fmaxf(cnt[node], 1.0f);
            size_t gb = (size_t)node * GATE3;

            float gr  = gi_acc[gb + o]       * inv + b_ih[o]       + lds_gh[row][o]       + b_hh[o];
            float gz  = gi_acc[gb + 128 + o] * inv + b_ih[128 + o] + lds_gh[row][128 + o] + b_hh[128 + o];
            float gin = gi_acc[gb + 256 + o] * inv + b_ih[256 + o];
            float ghn = lds_gh[row][256 + o] + b_hh[256 + o];

            float rg = 1.0f / (1.0f + __expf(-gr));
            float zg = 1.0f / (1.0f + __expf(-gz));
            float ng = tanhf(gin + rg * ghn);
            float h  = memory[(size_t)node * MEM_D + o];
            out_mem[(size_t)node * MEM_D + o] = (1.0f - zg) * ng + zg * h;
        }
    }

    // inventory passthrough (float4 copy: 256 cols = 64 float4s per row)
    for (int i = tid; i < ROWS * 64; i += 256) {
        int row = i >> 6, c4 = (i & 63) * 4;
        int node = base + row;
        if (node < NNODES) {
            *(float4*)&out_mem[(size_t)node * MEM_D + 128 + c4] =
                *(const float4*)&memory[(size_t)node * MEM_D + 128 + c4];
        }
    }

    if (tid < ROWS) {
        int node = base + tid;
        if (node < NNODES) out_lu[node] = (float)lu_max[node];
    }
}

// ---------------- host launch ----------------
static inline size_t align_up(size_t x, size_t a) { return (x + a - 1) & ~(a - 1); }

extern "C" void kernel_launch(void* const* d_in, const int* in_sizes, int n_in,
                              void* d_out, int out_size, void* d_ws, size_t ws_size,
                              hipStream_t stream) {
    // inputs (setup_inputs order): n_id, src, dst, prod, t, raw_msg, memory,
    // last_update, w_time, b_time, w_ih, w_hh, b_ih, b_hh
    const int*   src         = (const int*)  d_in[1];
    const int*   dst         = (const int*)  d_in[2];
    const int*   prod        = (const int*)  d_in[3];
    const int*   t           = (const int*)  d_in[4];
    const float* raw_msg     = (const float*)d_in[5];
    const float* memory      = (const float*)d_in[6];
    const int*   last_update = (const int*)  d_in[7];
    const float* w_time      = (const float*)d_in[8];
    const float* b_time      = (const float*)d_in[9];
    const float* w_ih        = (const float*)d_in[10];
    const float* w_hh        = (const float*)d_in[11];
    const float* b_ih        = (const float*)d_in[12];
    const float* b_hh        = (const float*)d_in[13];

    // workspace layout
    uint8_t* ws = (uint8_t*)d_ws;
    size_t off = 0;
    float* gi_acc = (float*)(ws + off);  off = align_up(off + (size_t)NNODES * GATE3 * 4, 256);
    float* cnt    = (float*)(ws + off);  off = align_up(off + (size_t)NNODES * 4, 256);
    int*   lu_max = (int*)  (ws + off);  off = align_up(off + (size_t)NNODES * 4, 256);
    __bf16* w_ih_bf = (__bf16*)(ws + off); off = align_up(off + (size_t)GATE3 * MSG_D * 2, 256);
    __bf16* w_hh_bf = (__bf16*)(ws + off); off = align_up(off + (size_t)GATE3 * STATE_D * 2, 256);
    (void)ws_size; (void)in_sizes; (void)n_in; (void)out_size;

    float* out_mem = (float*)d_out;
    float* out_lu  = out_mem + (size_t)NNODES * MEM_D;

    k_init<<<2048, 256, 0, stream>>>(gi_acc, cnt, lu_max);
    k_convert<<<512, 256, 0, stream>>>(w_ih, w_ih_bf, GATE3 * MSG_D);
    k_convert<<<64, 256, 0, stream>>>(w_hh, w_hh_bf, GATE3 * STATE_D);

    k_edge_gemm<<<(E_EDGES + ROWS - 1) / ROWS, 256, 0, stream>>>(
        src, dst, prod, t, raw_msg, memory, last_update, w_time, b_time,
        w_ih_bf, gi_acc, cnt, lu_max);

    k_gru<<<(NNODES + ROWS - 1) / ROWS, 256, 0, stream>>>(
        memory, w_hh_bf, b_ih, b_hh, gi_acc, cnt, lu_max, out_mem, out_lu);
}
